// NeuralStyleField_1726576854306
// MI455X (gfx1250) — matrix operand order repaired and checked by reading, not compiled, not run on hardware
//
#include <hip/hip_runtime.h>

typedef __attribute__((ext_vector_type(16))) _Float16 v16h;
typedef __attribute__((ext_vector_type(8)))  float    v8f;

#define NPTS   131072
#define MTILE  128           // rows per block (8 waves x 16 rows)
#define TPB    256
#define TAU_CONST (2.0f * 256.0f / 6000.0f)

#define CHUNK_HALVES 32768   // one staged chunk: 128 K x 256 N f16 = 64KB
#define N_CHUNKS     20      // W0(4) + {W1..W4,Wc0,Wc1,Wn0,Wn1}(2 each)

// ---- dynamic LDS layout (bytes) ----
// sW0   : 0      (+65536)  weight buffer A (async-DMA target)
// sW1   : 65536  (+65536)  weight buffer B (async-DMA target)
// sA    : 131072 (+65536)  activations, f16, A-fragment layout, 8 waves x 16x256
// sH    : 196608 (+65536)  saved trunk output h
// sBias : 262144 (+1024)
// sB3   : 263168 (+3072)   Fourier B, f32 [3][256]
// sW0x  : 266240 (+3072)   W0 rows 0..2, f32 [3][256]
// sX    : 269312 (+1536)   x tile, f32 [128][3]
// sHead : 270848 (+3072)   head final weights, f32
#define OFF_SW0   0
#define OFF_SW1   65536
#define OFF_SA    131072
#define OFF_SH    196608
#define OFF_BIAS  262144
#define OFF_B3    263168
#define OFF_W0X   266240
#define OFF_X     269312
#define OFF_HEAD  270848
#define SMEM_BYTES 273920

// ===========================================================================
// PREP KERNEL: convert all trunk weights f32 -> f16, swizzled into B-fragment
// layout, one 64KB chunk per block, into d_ws. Layout per chunk:
//   block = nt*4 + kl  (nt = n/16, kl = k_local/32)
//   lane  = (n&15) | (((k_in>>3)&1)<<4),  pos = (k_in&7) + ((k_in>>4)<<3)
// so each lane's B fragment is 16 contiguous halves in LDS after the copy.
// ===========================================================================
extern "C" __global__ void __launch_bounds__(TPB)
nsf_prep_weights(const float* __restrict__ W0, const float* __restrict__ W1,
                 const float* __restrict__ W2, const float* __restrict__ W3,
                 const float* __restrict__ W4, const float* __restrict__ Wc0,
                 const float* __restrict__ Wc1, const float* __restrict__ Wn0,
                 const float* __restrict__ Wn1, _Float16* __restrict__ wsW)
{
    const int chunk = blockIdx.x;          // 0..19
    const int tid   = threadIdx.x;

    const float* Wg;
    int k0;
    if (chunk < 4) {                       // W0 rows 3..514 (sin|cos block)
        Wg = W0 + 3 * 256;
        k0 = chunk * 128;
    } else {
        const float* tbl[8] = {W1, W2, W3, W4, Wc0, Wc1, Wn0, Wn1};
        Wg = tbl[(chunk - 4) >> 1];
        k0 = ((chunk - 4) & 1) * 128;
    }
    _Float16* dst = wsW + (size_t)chunk * CHUNK_HALVES;

    for (int kp = 0; kp < 64; ++kp) {
        int n   = tid;
        int kl2 = kp * 2;                  // even k_local
        float w0 = Wg[(size_t)(k0 + kl2) * 256 + n];
        float w1 = Wg[(size_t)(k0 + kl2 + 1) * 256 + n];
        int k_in   = kl2 & 31;
        int kl     = kl2 >> 5;
        int lane_w = (n & 15) | (((k_in >> 3) & 1) << 4);
        int pos    = (k_in & 7) + ((k_in >> 4) << 3);   // even
        int nt     = n >> 4;
        int off    = ((((nt << 2) | kl) * 32 + lane_w) << 4) + pos;
        union { _Float16 h[2]; unsigned u; } pk;
        pk.h[0] = (_Float16)w0; pk.h[1] = (_Float16)w1;
        *(unsigned*)(dst + off) = pk.u;
    }
}

// ===========================================================================
// MAIN KERNEL helpers
// ===========================================================================

// Issue async DMA of one 64KB weight chunk (pre-swizzled f16 in ws) into the
// LDS buffer selected by chunk parity. 16 x global_load_async_to_lds_b128 per
// thread (256B slice each); tracked by ASYNCcnt (16 per chunk per wave).
__device__ __forceinline__ void issue_chunk(const _Float16* __restrict__ wsW,
                                            int c, char* smem, int tid)
{
    const char* g = (const char*)(wsW + (size_t)c * CHUNK_HALVES) + tid * 256;
    unsigned lds  = (unsigned)(uintptr_t)(smem + ((c & 1) ? OFF_SW1 : OFF_SW0))
                  + (unsigned)(tid * 256);
#pragma unroll
    for (int i = 0; i < 16; ++i) {
        asm volatile("global_load_async_to_lds_b128 %0, %1, off offset:%2"
                     :: "v"(lds), "v"(g), "n"(i * 16) : "memory");
    }
}

__device__ __forceinline__ void wait_async16() {
    asm volatile("s_wait_asynccnt 16" ::: "memory");
}
__device__ __forceinline__ void wait_async0() {
    asm volatile("s_wait_asynccnt 0" ::: "memory");
}

// 4 k-steps x 16 n-tiles of v_wmma_f32_16x16x32_f16 against a staged chunk.
__device__ __forceinline__ void mm_chunk(v8f acc[16], const _Float16* sW,
                                         const _Float16* sAw, int lane, int ksBase)
{
#pragma unroll
    for (int kl = 0; kl < 4; ++kl) {
        v16h a = *(const v16h*)(sAw + (((ksBase + kl) * 32 + lane) << 4));
#pragma unroll
        for (int nt = 0; nt < 16; ++nt) {
            v16h b = *(const v16h*)(sW + (((nt * 4 + kl) * 32 + lane) << 4));
            acc[nt] = __builtin_amdgcn_wmma_f32_16x16x32_f16(
                false, a, false, b, (short)0, acc[nt], false, false);
        }
    }
}

// bias + ReLU, write f16 activations into A-fragment layout.
__device__ __forceinline__ void epilogue_relu(v8f acc[16], const float* sBias,
                                              _Float16* dst, int lane)
{
#pragma unroll
    for (int nt = 0; nt < 16; ++nt) {
        int n      = nt * 16 + (lane & 15);
        float bias = sBias[n];
        int ks2    = n >> 5;
        int k_in   = n & 31;
        int group  = k_in >> 3;
        int lane_b = (group & 1) << 4;
        int pos    = (k_in & 7) + ((group >> 1) << 3);
#pragma unroll
        for (int j = 0; j < 8; ++j) {
            int Mloc = j + ((lane >> 4) << 3);
            float v  = fmaxf(acc[nt][j] + bias, 0.0f);
            dst[((ks2 * 32 + Mloc + lane_b) << 4) + pos] = (_Float16)v;
        }
    }
}

// One 256x256 Linear + ReLU layer consuming chunks C0, C0+1 (already in
// flight) and prefetching C0+2, C0+3 under the WMMA work.
template <int C0, bool LAST>
__device__ __forceinline__ void layer_async(const float* __restrict__ bg,
                                            const _Float16* __restrict__ wsW,
                                            char* smem, float* sBias,
                                            const _Float16* src, _Float16* dst,
                                            int tid, int lane)
{
    const _Float16* sWa = (const _Float16*)(smem + ((C0 & 1) ? OFF_SW1 : OFF_SW0));
    const _Float16* sWb = (const _Float16*)(smem + ((C0 & 1) ? OFF_SW0 : OFF_SW1));

    wait_async16();                 // chunk C0 resident (C0+1 may be in flight)
    __syncthreads();
    sBias[tid] = bg[tid];

    v8f acc[16];
    {
        v8f vz = {0.f,0.f,0.f,0.f,0.f,0.f,0.f,0.f};
#pragma unroll
        for (int nt = 0; nt < 16; ++nt) acc[nt] = vz;
    }

    mm_chunk(acc, sWa, src, lane, 0);
    __syncthreads();                // all waves done with buffer A
    if (C0 + 2 < N_CHUNKS) issue_chunk(wsW, C0 + 2, smem, tid);

    if (LAST) wait_async0(); else wait_async16();   // chunk C0+1 resident
    __syncthreads();
    mm_chunk(acc, sWb, src, lane, 4);
    __syncthreads();                // all waves done with buffer B
    if (C0 + 3 < N_CHUNKS) issue_chunk(wsW, C0 + 3, smem, tid);

    epilogue_relu(acc, sBias, dst, lane);   // wave-private dst
}

// sin(pass0)/cos(pass1) Fourier features * alpha into the wave's A buffer.
__device__ __forceinline__ void fill_features(int pass, const float* sB3,
                                              const float* sX, _Float16* sAw,
                                              int lane, int wave, float t_over_tau)
{
    int r  = wave * 16 + (lane & 15);
    float x0 = sX[r * 3 + 0], x1 = sX[r * 3 + 1], x2 = sX[r * 3 + 2];
    int hi = lane >> 4;
#pragma unroll
    for (int ks = 0; ks < 8; ++ks) {
        v16h frag;
#pragma unroll
        for (int i = 0; i < 16; ++i) {
            int group = ((i >> 3) << 1) | hi;
            int c     = ks * 32 + group * 8 + (i & 7);
            float proj = 6.28318530718f *
                (x0 * sB3[c] + x1 * sB3[256 + c] + x2 * sB3[512 + c]);
            float av = t_over_tau - (float)c;
            av = fminf(fmaxf(av, 0.0f), 1.0f);
            float v = (pass == 0 ? __sinf(proj) : __cosf(proj)) * av;
            frag[i] = (_Float16)v;
        }
        *(v16h*)(sAw + ((ks * 32 + lane) << 4)) = frag;
    }
}

// ===========================================================================
// MAIN KERNEL
// ===========================================================================
extern "C" __global__ void __launch_bounds__(TPB, 1)
nsf_fused_wmma(const float* __restrict__ x, const float* __restrict__ Bf,
               const int* __restrict__ tptr,
               const float* __restrict__ W0, const float* __restrict__ b0,
               const float* __restrict__ b1, const float* __restrict__ b2,
               const float* __restrict__ b3, const float* __restrict__ b4,
               const float* __restrict__ bc0, const float* __restrict__ bc1,
               const float* __restrict__ Wc2, const float* __restrict__ bc2,
               const float* __restrict__ bn0, const float* __restrict__ bn1,
               const float* __restrict__ Wn2, const float* __restrict__ bn2,
               const _Float16* __restrict__ wsW,
               float* __restrict__ out)
{
    extern __shared__ char smem[];
    _Float16* sA    = (_Float16*)(smem + OFF_SA);
    _Float16* sH    = (_Float16*)(smem + OFF_SH);
    float*    sBias = (float*)(smem + OFF_BIAS);
    float*    sB3   = (float*)(smem + OFF_B3);
    float*    sW0x  = (float*)(smem + OFF_W0X);
    float*    sX    = (float*)(smem + OFF_X);
    float*    sHead = (float*)(smem + OFF_HEAD);

    const int tid     = threadIdx.x;
    const int lane    = tid & 31;
    const int wave    = tid >> 5;
    const int rowBase = blockIdx.x * MTILE;
    _Float16* sAw = sA + wave * 4096;
    _Float16* sHw = sH + wave * 4096;

    // kick off DMA of the first two weight chunks immediately
    issue_chunk(wsW, 0, smem, tid);
    issue_chunk(wsW, 1, smem, tid);

    const float t_over_tau = (float)(*tptr) / TAU_CONST;

    for (int i = tid; i < 768; i += TPB) sB3[i]  = Bf[i];
    for (int i = tid; i < 384; i += TPB) sX[i]   = x[(size_t)rowBase * 3 + i];
    for (int i = tid; i < 768; i += TPB) sW0x[i] = W0[i];   // rows 0..2
    sBias[tid] = b0[tid];
    __syncthreads();

    // =================== Layer 0: feat(515) @ W0, chunks 0..3 ===============
    v8f acc[16];
    {
        v8f vz = {0.f,0.f,0.f,0.f,0.f,0.f,0.f,0.f};
#pragma unroll
        for (int nt = 0; nt < 16; ++nt) acc[nt] = vz;
    }
    const _Float16* sWbuf0 = (const _Float16*)(smem + OFF_SW0);
    const _Float16* sWbuf1 = (const _Float16*)(smem + OFF_SW1);

    fill_features(0, sB3, sX, sAw, lane, wave, t_over_tau);   // sin
    wait_async16(); __syncthreads();
    mm_chunk(acc, sWbuf0, sAw, lane, 0);                      // chunk 0
    __syncthreads(); issue_chunk(wsW, 2, smem, tid);
    wait_async16(); __syncthreads();
    mm_chunk(acc, sWbuf1, sAw, lane, 4);                      // chunk 1
    __syncthreads(); issue_chunk(wsW, 3, smem, tid);

    fill_features(1, sB3, sX, sAw, lane, wave, t_over_tau);   // cos
    wait_async16(); __syncthreads();
    mm_chunk(acc, sWbuf0, sAw, lane, 0);                      // chunk 2
    __syncthreads(); issue_chunk(wsW, 4, smem, tid);
    wait_async16(); __syncthreads();
    mm_chunk(acc, sWbuf1, sAw, lane, 4);                      // chunk 3
    __syncthreads(); issue_chunk(wsW, 5, smem, tid);

    // epilogue: + b0 + x @ W0[0:3] correction, ReLU, write to sA
    {
        float xr0[8], xr1[8], xr2[8];
#pragma unroll
        for (int j = 0; j < 8; ++j) {
            int Mloc = j + ((lane >> 4) << 3);
            xr0[j] = sX[(wave * 16 + Mloc) * 3 + 0];
            xr1[j] = sX[(wave * 16 + Mloc) * 3 + 1];
            xr2[j] = sX[(wave * 16 + Mloc) * 3 + 2];
        }
#pragma unroll
        for (int nt = 0; nt < 16; ++nt) {
            int n      = nt * 16 + (lane & 15);
            float bias = sBias[n];
            float wa = sW0x[n], wb = sW0x[256 + n], wc = sW0x[512 + n];
            int ks2    = n >> 5;
            int k_in   = n & 31;
            int group  = k_in >> 3;
            int lane_b = (group & 1) << 4;
            int pos    = (k_in & 7) + ((group >> 1) << 3);
#pragma unroll
            for (int j = 0; j < 8; ++j) {
                int Mloc = j + ((lane >> 4) << 3);
                float v = acc[nt][j] + bias + xr0[j] * wa + xr1[j] * wb + xr2[j] * wc;
                v = fmaxf(v, 0.0f);
                sAw[((ks2 * 32 + Mloc + lane_b) << 4) + pos] = (_Float16)v;
            }
        }
    }

    // =================== trunk layers 1..4 (chunks 4..11) ===================
    layer_async< 4, false>(b1, wsW, smem, sBias, sAw, sAw, tid, lane);
    layer_async< 6, false>(b2, wsW, smem, sBias, sAw, sAw, tid, lane);
    layer_async< 8, false>(b3, wsW, smem, sBias, sAw, sAw, tid, lane);
    layer_async<10, false>(b4, wsW, smem, sBias, sAw, sHw, tid, lane);  // save h

    // =================== color head (chunks 12..15) ===================
    layer_async<12, false>(bc0, wsW, smem, sBias, sHw, sAw, tid, lane);
    layer_async<14, false>(bc1, wsW, smem, sBias, sAw, sAw, tid, lane);
    __syncthreads();
    for (int i = tid; i < 768; i += TPB) sHead[i] = Wc2[i];   // [256][3]
    __syncthreads();
    {
        float p0 = 0.f, p1 = 0.f, p2 = 0.f;
        int hi = lane >> 4;
#pragma unroll
        for (int ks = 0; ks < 8; ++ks) {
            v16h av = *(const v16h*)(sAw + ((ks * 32 + lane) << 4));
#pragma unroll
            for (int i = 0; i < 16; ++i) {
                int group = ((i >> 3) << 1) | hi;
                int k     = ks * 32 + group * 8 + (i & 7);
                float hv  = (float)av[i];
                p0 += hv * sHead[k * 3 + 0];
                p1 += hv * sHead[k * 3 + 1];
                p2 += hv * sHead[k * 3 + 2];
            }
        }
        p0 += __shfl_xor(p0, 16);
        p1 += __shfl_xor(p1, 16);
        p2 += __shfl_xor(p2, 16);
        if (lane < 16) {
            int row = rowBase + wave * 16 + lane;
            out[(size_t)row * 3 + 0] = tanhf(p0 + bc2[0]) * 0.5f;
            out[(size_t)row * 3 + 1] = tanhf(p1 + bc2[1]) * 0.5f;
            out[(size_t)row * 3 + 2] = tanhf(p2 + bc2[2]) * 0.5f;
        }
    }

    // =================== displacement head (chunks 16..19) ===================
    layer_async<16, false>(bn0, wsW, smem, sBias, sHw, sAw, tid, lane);
    layer_async<18, true >(bn1, wsW, smem, sBias, sAw, sAw, tid, lane);
    __syncthreads();
    for (int i = tid; i < 256; i += TPB) sHead[i] = Wn2[i];   // [256][1]
    __syncthreads();
    {
        float p0 = 0.f;
        int hi = lane >> 4;
#pragma unroll
        for (int ks = 0; ks < 8; ++ks) {
            v16h av = *(const v16h*)(sAw + ((ks * 32 + lane) << 4));
#pragma unroll
            for (int i = 0; i < 16; ++i) {
                int group = ((i >> 3) << 1) | hi;
                int k     = ks * 32 + group * 8 + (i & 7);
                p0 += (float)av[i] * sHead[k];
            }
        }
        p0 += __shfl_xor(p0, 16);
        if (lane < 16) {
            int row = rowBase + wave * 16 + lane;
            out[(size_t)3 * NPTS + row] = tanhf(p0 + bn2[0]) * 0.1f;
        }
    }
}

// ===========================================================================
extern "C" void kernel_launch(void* const* d_in, const int* in_sizes, int n_in,
                              void* d_out, int out_size, void* d_ws, size_t ws_size,
                              hipStream_t stream)
{
    (void)in_sizes; (void)n_in; (void)out_size; (void)ws_size;
    const float* x   = (const float*)d_in[0];
    const float* Bf  = (const float*)d_in[1];
    const int*   t   = (const int*)  d_in[2];
    const float* W0  = (const float*)d_in[3];
    const float* b0  = (const float*)d_in[4];
    const float* W1  = (const float*)d_in[5];
    const float* b1  = (const float*)d_in[6];
    const float* W2  = (const float*)d_in[7];
    const float* b2  = (const float*)d_in[8];
    const float* W3  = (const float*)d_in[9];
    const float* b3  = (const float*)d_in[10];
    const float* W4  = (const float*)d_in[11];
    const float* b4  = (const float*)d_in[12];
    const float* Wc0 = (const float*)d_in[13];
    const float* bc0 = (const float*)d_in[14];
    const float* Wc1 = (const float*)d_in[15];
    const float* bc1 = (const float*)d_in[16];
    const float* Wc2 = (const float*)d_in[17];
    const float* bc2 = (const float*)d_in[18];
    const float* Wn0 = (const float*)d_in[19];
    const float* bn0 = (const float*)d_in[20];
    const float* Wn1 = (const float*)d_in[21];
    const float* bn1 = (const float*)d_in[22];
    const float* Wn2 = (const float*)d_in[23];
    const float* bn2 = (const float*)d_in[24];
    float* out = (float*)d_out;
    _Float16* wsW = (_Float16*)d_ws;      // 20 x 64KB swizzled f16 weights

    // 1) one-time (per launch) weight convert+swizzle into workspace
    nsf_prep_weights<<<dim3(N_CHUNKS), dim3(TPB), 0, stream>>>(
        W0, W1, W2, W3, W4, Wc0, Wc1, Wn0, Wn1, wsW);

    // 2) fused MLP, weights streamed LDS-ward via async DMA
    nsf_fused_wmma<<<dim3(NPTS / MTILE), dim3(TPB), SMEM_BYTES, stream>>>(
        x, Bf, t, W0, b0, b1, b2, b3, b4, bc0, bc1, Wc2, bc2,
        bn0, bn1, Wn2, bn2, wsW, out);
}
